// SingleLayerGAT_90202903151253
// MI455X (gfx1250) — matrix operand order repaired and checked by reading, not compiled, not run on hardware
//
#include <hip/hip_runtime.h>
#include <hip/hip_bf16.h>

typedef __attribute__((ext_vector_type(16))) _Float16 v16h;
typedef __attribute__((ext_vector_type(8)))  _Float16 v8h;
typedef __attribute__((ext_vector_type(8)))  float    v8f;

// ---------------- device helpers ----------------
__device__ __forceinline__ float gelu_f(float x) {
    const float k0 = 0.7978845608028654f, k1 = 0.044715f;
    float x3 = x * x * x;
    return 0.5f * x * (1.0f + tanhf(k0 * (x + k1 * x3)));
}
__device__ __forceinline__ unsigned f2ord(float f) {
    unsigned u = __float_as_uint(f);
    return (u & 0x80000000u) ? ~u : (u | 0x80000000u);
}
__device__ __forceinline__ float ord2f(unsigned u) {
    return (u & 0x80000000u) ? __uint_as_float(u & 0x7fffffffu) : __uint_as_float(~u);
}
__device__ __forceinline__ v16h pack16(const _Float16* p0, const _Float16* p1) {
    v8h a0 = *(const v8h*)p0;
    v8h a1 = *(const v8h*)p1;
    v16h a;
#pragma unroll
    for (int j = 0; j < 8; j++) { a[j] = a0[j]; a[8 + j] = a1[j]; }
    return a;
}

// ---------------- elementwise utility kernels ----------------
__global__ void k_f2h(const float* __restrict__ s, _Float16* __restrict__ d, int n) {
    int t = blockIdx.x * blockDim.x + threadIdx.x;
    if (t < n) d[t] = (_Float16)s[t];
}
// dst[n*K+k] = (f16) src[k*N+n]   (weight transpose + downconvert)
__global__ void k_transpose_h(const float* __restrict__ src, _Float16* __restrict__ dst,
                              int K, int N) {
    int t = blockIdx.x * blockDim.x + threadIdx.x;
    if (t >= K * N) return;
    int n = t / K, k = t % K;
    dst[t] = (_Float16)src[(size_t)k * N + n];
}
__global__ void k_fill_u32(unsigned* __restrict__ p, unsigned v, int n) {
    int t = blockIdx.x * blockDim.x + threadIdx.x;
    if (t < n) p[t] = v;
}
__global__ void k_fill_f32(float* __restrict__ p, float v, int n) {
    int t = blockIdx.x * blockDim.x + threadIdx.x;
    if (t < n) p[t] = v;
}

// ---------------- WMMA GEMM: C[M,N] = act(A[M,K] @ Bt[N,K]^T + bias) ----------------
// A: f16 row-major (optionally gathered by rowIdx), Bt: f16 row-major (B transposed).
// One wave -> one 16(M) x 64(N) strip = 4 WMMA tiles sharing the A fragment.
// K multiple of 32, N multiple of 64. M arbitrary (row-clamped; padded rows not stored).
__global__ void k_gemm_wmma(const _Float16* __restrict__ A,
                            const _Float16* __restrict__ Bt,
                            const float* __restrict__ bias,
                            float* __restrict__ Cf, _Float16* __restrict__ Ch,
                            const int* __restrict__ rowIdx,
                            int M, int N, int K, int act) {
    int wave = (blockIdx.x * blockDim.x + threadIdx.x) >> 5;
    int lane = threadIdx.x & 31;
    int tilesQ = N >> 6;                       // groups of 4 N-tiles
    int tilesM = (M + 15) >> 4;
    if (wave >= tilesM * tilesQ) return;       // wave-uniform exit (EXEC stays full)
    int tm = wave / tilesQ, tq = wave % tilesQ;
    int r = lane & 15, g = lane >> 4;

    int m = tm * 16 + r;
    int mc = (m < M) ? m : (M - 1);            // clamp: padded D rows are never stored
    int msrc = rowIdx ? rowIdx[mc] : mc;
    const _Float16* arow = A + (size_t)msrc * K;
    const _Float16* brow = Bt + (size_t)(tq * 64 + r) * K;   // q-th tile at +16*K

    v8f c0, c1, c2, c3;
#pragma unroll
    for (int i = 0; i < 8; i++) { c0[i] = 0.f; c1[i] = 0.f; c2[i] = 0.f; c3[i] = 0.f; }

    for (int k0 = 0; k0 < K; k0 += 32) {
        if (k0 + 64 < K) __builtin_prefetch(arow + k0 + 64, 0, 0);
        // A fragment: halves j -> K = k0 + 16g + j
        v16h a = pack16(arow + k0 + g * 16, arow + k0 + g * 16 + 8);
        // B fragments: halves j -> K = k0 + 16g + j, lane column = 16q + r
        v16h b0 = pack16(brow + k0 + g * 16,              brow + k0 + g * 16 + 8);
        v16h b1 = pack16(brow + (size_t)16 * K + k0 + g * 16, brow + (size_t)16 * K + k0 + g * 16 + 8);
        v16h b2 = pack16(brow + (size_t)32 * K + k0 + g * 16, brow + (size_t)32 * K + k0 + g * 16 + 8);
        v16h b3 = pack16(brow + (size_t)48 * K + k0 + g * 16, brow + (size_t)48 * K + k0 + g * 16 + 8);
        c0 = __builtin_amdgcn_wmma_f32_16x16x32_f16(false, a, false, b0, (short)0, c0, false, false);
        c1 = __builtin_amdgcn_wmma_f32_16x16x32_f16(false, a, false, b1, (short)0, c1, false, false);
        c2 = __builtin_amdgcn_wmma_f32_16x16x32_f16(false, a, false, b2, (short)0, c2, false, false);
        c3 = __builtin_amdgcn_wmma_f32_16x16x32_f16(false, a, false, b3, (short)0, c3, false, false);
    }

#pragma unroll
    for (int q = 0; q < 4; q++) {
        v8f c = (q == 0) ? c0 : (q == 1) ? c1 : (q == 2) ? c2 : c3;
        int col = tq * 64 + q * 16 + r;
        float bcol = bias ? bias[col] : 0.0f;
#pragma unroll
        for (int v = 0; v < 8; v++) {
            int mo = tm * 16 + g * 8 + v;       // D layout: M = 8*g + v, N = lane%16
            if (mo < M) {
                float val = c[v] + bcol;
                if (act == 1) val = gelu_f(val);
                size_t o = (size_t)mo * N + col;
                if (Cf) Cf[o] = val;
                if (Ch) Ch[o] = (_Float16)val;
            }
        }
    }
}

// ---------------- attention folding: w[k*4+h] = sum_c lin[k, h*64+c]*att[h*64+c] ----------------
__global__ void k_att_w(const float* __restrict__ lin, const float* __restrict__ att,
                        float* __restrict__ w) {
    int t = blockIdx.x * blockDim.x + threadIdx.x;
    if (t >= 256 * 4) return;
    int k = t >> 2, h = t & 3;
    float s = 0.f;
    for (int c = 0; c < 64; c++) s += lin[(size_t)k * 256 + h * 64 + c] * att[h * 64 + c];
    w[k * 4 + h] = s;
}
// a_d[n,h] = sum_k h_gene[n,k] * w_ad[k,h]
__global__ void k_ad(const float* __restrict__ hg, const float* __restrict__ w,
                     float* __restrict__ ad, int Nrow) {
    int n = blockIdx.x * blockDim.x + threadIdx.x;
    if (n >= Nrow) return;
    const float* row = hg + (size_t)n * 256;
    float a0 = 0.f, a1 = 0.f, a2 = 0.f, a3 = 0.f;
    for (int k = 0; k < 256; k++) {
        float x = row[k];
        const float* wk = w + k * 4;
        a0 += x * wk[0]; a1 += x * wk[1]; a2 += x * wk[2]; a3 += x * wk[3];
    }
    float* o = ad + (size_t)n * 4;
    o[0] = a0; o[1] = a1; o[2] = a2; o[3] = a3;
}
// a_s[n,h] = sum_c hs[n, h*64+c] * att_src[h*64+c]
__global__ void k_as(const float* __restrict__ hs, const float* __restrict__ att,
                     float* __restrict__ as_, int Nm) {
    int t = blockIdx.x * blockDim.x + threadIdx.x;
    if (t >= Nm * 4) return;
    int n = t >> 2, h = t & 3;
    float s = 0.f;
    for (int c = 0; c < 64; c++) s += hs[(size_t)n * 256 + h * 64 + c] * att[h * 64 + c];
    as_[t] = s;
}

// ---------------- edge passes ----------------
__global__ void k_edge_max(const int* __restrict__ src, const int* __restrict__ dst,
                           const float* __restrict__ as_, const float* __restrict__ ad,
                           float* __restrict__ ebuf, unsigned* __restrict__ mord, int E) {
    int e = blockIdx.x * blockDim.x + threadIdx.x;
    if (e >= E) return;
    int s = src[e], d = dst[e];
#pragma unroll
    for (int h = 0; h < 4; h++) {
        float v = as_[(size_t)s * 4 + h] + ad[(size_t)d * 4 + h];
        v = (v > 0.f) ? v : 0.2f * v;                 // LeakyReLU(0.2)
        ebuf[(size_t)e * 4 + h] = v;
        atomicMax(mord + (size_t)d * 4 + h, f2ord(v));
    }
}
__global__ void k_edge_exp(const int* __restrict__ dst, float* __restrict__ ebuf,
                           const unsigned* __restrict__ mord, float* __restrict__ ssum, int E) {
    int e = blockIdx.x * blockDim.x + threadIdx.x;
    if (e >= E) return;
    int d = dst[e];
#pragma unroll
    for (int h = 0; h < 4; h++) {
        float m = ord2f(mord[(size_t)d * 4 + h]);
        float ex = expf(ebuf[(size_t)e * 4 + h] - m);
        ebuf[(size_t)e * 4 + h] = ex;
        atomicAdd(ssum + (size_t)d * 4 + h, ex);
    }
}
__global__ void k_edge_alpha(const int* __restrict__ dst, float* __restrict__ ebuf,
                             const float* __restrict__ ssum, int E) {
    int e = blockIdx.x * blockDim.x + threadIdx.x;
    if (e >= E) return;
    int d = dst[e];
#pragma unroll
    for (int h = 0; h < 4; h++)
        ebuf[(size_t)e * 4 + h] /= (ssum[(size_t)d * 4 + h] + 1e-16f);
}
__global__ void k_edge_scatter(const int* __restrict__ src, const int* __restrict__ dst,
                               const float* __restrict__ alpha, const float* __restrict__ hs,
                               float* __restrict__ acc, int E) {
    long t = (long)blockIdx.x * blockDim.x + threadIdx.x;
    if (t >= (long)E * 64) return;
    int e = (int)(t >> 6), c = (int)(t & 63);
    int s = src[e], d = dst[e];
#pragma unroll
    for (int h = 0; h < 4; h++) {
        float al = alpha[(size_t)e * 4 + h];
        atomicAdd(acc + (size_t)d * 256 + h * 64 + c,
                  hs[(size_t)s * 256 + h * 64 + c] * al);
    }
}

// ---------------- residual + GELU + LayerNorm (wave per row of 256) ----------------
__global__ void k_post_ln(const float* __restrict__ acc, const float* __restrict__ bias,
                          const float* __restrict__ hres, const float* __restrict__ gamma,
                          const float* __restrict__ beta, _Float16* __restrict__ out, int Nrow) {
    int wid = (blockIdx.x * blockDim.x + threadIdx.x) >> 5;
    int lane = threadIdx.x & 31;
    if (wid >= Nrow) return;
    const float* pa = acc + (size_t)wid * 256;
    const float* ph = hres + (size_t)wid * 256;
    float x[8], s = 0.f;
#pragma unroll
    for (int i = 0; i < 8; i++) {
        int c = lane * 8 + i;
        float v = gelu_f(pa[c] + bias[c] + ph[c]);
        x[i] = v; s += v;
    }
#pragma unroll
    for (int m = 16; m; m >>= 1) s += __shfl_xor(s, m, 32);
    float mu = s * (1.0f / 256.0f);
    float var = 0.f;
#pragma unroll
    for (int i = 0; i < 8; i++) { float d = x[i] - mu; var += d * d; }
#pragma unroll
    for (int m = 16; m; m >>= 1) var += __shfl_xor(var, m, 32);
    float r = rsqrtf(var * (1.0f / 256.0f) + 1e-5f);
#pragma unroll
    for (int i = 0; i < 8; i++) {
        int c = lane * 8 + i;
        out[(size_t)wid * 256 + c] = (_Float16)((x[i] - mu) * r * gamma[c] + beta[c]);
    }
}

// ---------------- final bilinear dot ----------------
__global__ void k_final(const float* __restrict__ t, const float* __restrict__ drug,
                        const int* __restrict__ didx, const float* __restrict__ bbil,
                        float* __restrict__ out, int B) {
    int b = blockIdx.x * blockDim.x + threadIdx.x;
    if (b >= B) return;
    const float* tv = t + (size_t)b * 128;
    const float* dv = drug + (size_t)didx[b] * 128;
    float s = 0.f;
    for (int c = 0; c < 128; c++) s += tv[c] * dv[c];
    out[b] = s + bbil[0];
}

// ---------------- host ----------------
static inline int cdiv(long a, long b) { return (int)((a + b - 1) / b); }

extern "C" void kernel_launch(void* const* d_in, const int* in_sizes, int n_in,
                              void* d_out, int out_size, void* d_ws, size_t ws_size,
                              hipStream_t stream) {
    const int Ng = 50000, Nm = 5000, E = 300000, B = 16384, Ndr = 1000;
    const int gd = 512, md = 256, dd = 512, hid = 256, od = 128;

    const float* gene_x  = (const float*)d_in[0];
    const float* mech_x  = (const float*)d_in[1];
    const float* drug_x  = (const float*)d_in[2];
    const float* Wg      = (const float*)d_in[3];
    const float* bg      = (const float*)d_in[4];
    const float* Wm      = (const float*)d_in[5];
    const float* bm      = (const float*)d_in[6];
    const float* lin     = (const float*)d_in[7];
    const float* att_src = (const float*)d_in[8];
    const float* att_dst = (const float*)d_in[9];
    const float* bias_mg = (const float*)d_in[10];
    const float* Wo      = (const float*)d_in[11];
    const float* bo      = (const float*)d_in[12];
    const float* gamma   = (const float*)d_in[13];
    const float* beta    = (const float*)d_in[14];
    const float* Wd1     = (const float*)d_in[15];
    const float* bd1     = (const float*)d_in[16];
    const float* Wd2     = (const float*)d_in[17];
    const float* bd2     = (const float*)d_in[18];
    const float* Wbil    = (const float*)d_in[19];
    const float* bbil    = (const float*)d_in[20];
    const int* ei_src    = (const int*)d_in[21];
    const int* ei_dst    = (const int*)d_in[22];
    const int* gene_idx  = (const int*)d_in[23];
    const int* drug_idx  = (const int*)d_in[24];
    float* out = (float*)d_out;

    char* ws = (char*)d_ws;
    size_t off = 0;
    auto alloc = [&](size_t bytes) -> void* {
        void* p = ws + off;
        off = (off + bytes + 255) & ~(size_t)255;
        return p;
    };

    _Float16* geneA = (_Float16*)alloc((size_t)Ng * gd * 2);
    _Float16* mechA = (_Float16*)alloc((size_t)Nm * md * 2);
    _Float16* drugA = (_Float16*)alloc((size_t)Ndr * dd * 2);
    _Float16* WgT   = (_Float16*)alloc((size_t)gd * hid * 2);
    _Float16* WmT   = (_Float16*)alloc((size_t)md * hid * 2);
    _Float16* linT  = (_Float16*)alloc((size_t)hid * hid * 2);
    _Float16* WoT   = (_Float16*)alloc((size_t)hid * od * 2);
    _Float16* Wd1T  = (_Float16*)alloc((size_t)dd * hid * 2);
    _Float16* Wd2T  = (_Float16*)alloc((size_t)hid * od * 2);
    _Float16* WbilT = (_Float16*)alloc((size_t)od * od * 2);
    float*    h_gene = (float*)alloc((size_t)Ng * hid * 4);
    _Float16* h_mech = (_Float16*)alloc((size_t)Nm * hid * 2);
    float*    hs     = (float*)alloc((size_t)Nm * hid * 4);
    float*    w_as   = (float*)alloc(256 * 4 * 4);
    float*    w_ad   = (float*)alloc(256 * 4 * 4);
    float*    a_s    = (float*)alloc((size_t)Nm * 4 * 4);
    float*    a_d    = (float*)alloc((size_t)Ng * 4 * 4);
    unsigned* mord   = (unsigned*)alloc((size_t)Ng * 4 * 4);
    float*    ssum   = (float*)alloc((size_t)Ng * 4 * 4);
    float*    ebuf   = (float*)alloc((size_t)E * 4 * 4);
    float*    acc    = (float*)alloc((size_t)Ng * hid * 4);
    _Float16* LNh    = (_Float16*)alloc((size_t)Ng * hid * 2);
    _Float16* geneGh = (_Float16*)alloc((size_t)B * od * 2);
    float*    tbuf   = (float*)alloc((size_t)B * od * 4);
    _Float16* d1h    = (_Float16*)alloc((size_t)Ndr * hid * 2);
    float*    drugE  = (float*)alloc((size_t)Ndr * od * 4);
    (void)ws_size; (void)n_in; (void)in_sizes; (void)out_size;

    const int TB = 256;
    // f32 -> f16 activations
    k_f2h<<<cdiv((long)Ng * gd, TB), TB, 0, stream>>>(gene_x, geneA, Ng * gd);
    k_f2h<<<cdiv((long)Nm * md, TB), TB, 0, stream>>>(mech_x, mechA, Nm * md);
    k_f2h<<<cdiv((long)Ndr * dd, TB), TB, 0, stream>>>(drug_x, drugA, Ndr * dd);
    // weight transposes (+downconvert)
    k_transpose_h<<<cdiv((long)gd * hid, TB), TB, 0, stream>>>(Wg, WgT, gd, hid);
    k_transpose_h<<<cdiv((long)md * hid, TB), TB, 0, stream>>>(Wm, WmT, md, hid);
    k_transpose_h<<<cdiv((long)hid * hid, TB), TB, 0, stream>>>(lin, linT, hid, hid);
    k_transpose_h<<<cdiv((long)hid * od, TB), TB, 0, stream>>>(Wo, WoT, hid, od);
    k_transpose_h<<<cdiv((long)dd * hid, TB), TB, 0, stream>>>(Wd1, Wd1T, dd, hid);
    k_transpose_h<<<cdiv((long)hid * od, TB), TB, 0, stream>>>(Wd2, Wd2T, hid, od);
    k_transpose_h<<<cdiv((long)od * od, TB), TB, 0, stream>>>(Wbil, WbilT, od, od);

    auto gemm = [&](const _Float16* A, const _Float16* Bt, const float* bias,
                    float* Cf, _Float16* Ch, const int* ridx, int M, int N, int K, int act) {
        long waves = (long)((M + 15) / 16) * (N / 64);
        k_gemm_wmma<<<cdiv(waves, 4), 128, 0, stream>>>(A, Bt, bias, Cf, Ch, ridx, M, N, K, act);
    };

    // 1) h_gene = gelu(gene_x @ Wg + bg)  [f32]
    gemm(geneA, WgT, bg, h_gene, nullptr, nullptr, Ng, hid, gd, 1);
    // 2) h_mech = gelu(mech_x @ Wm + bm)  [f16]
    gemm(mechA, WmT, bm, nullptr, h_mech, nullptr, Nm, hid, md, 1);
    // 3) hs = h_mech @ lin  [f32]
    gemm(h_mech, linT, nullptr, hs, nullptr, nullptr, Nm, hid, hid, 0);
    // folded attention weights + logits
    k_att_w<<<cdiv(1024, TB), TB, 0, stream>>>(lin, att_src, w_as);
    k_att_w<<<cdiv(1024, TB), TB, 0, stream>>>(lin, att_dst, w_ad);
    k_ad<<<cdiv(Ng, TB), TB, 0, stream>>>(h_gene, w_ad, a_d, Ng);
    k_as<<<cdiv((long)Nm * 4, TB), TB, 0, stream>>>(hs, att_src, a_s, Nm);
    // segment softmax buffers
    k_fill_u32<<<cdiv((long)Ng * 4, TB), TB, 0, stream>>>(mord, 0x007FFFFFu, Ng * 4); // ord(-inf)
    k_fill_f32<<<cdiv((long)Ng * 4, TB), TB, 0, stream>>>(ssum, 0.0f, Ng * 4);
    k_fill_f32<<<cdiv((long)Ng * hid, TB), TB, 0, stream>>>(acc, 0.0f, Ng * hid);
    // edge passes
    k_edge_max<<<cdiv(E, TB), TB, 0, stream>>>(ei_src, ei_dst, a_s, a_d, ebuf, mord, E);
    k_edge_exp<<<cdiv(E, TB), TB, 0, stream>>>(ei_dst, ebuf, mord, ssum, E);
    k_edge_alpha<<<cdiv(E, TB), TB, 0, stream>>>(ei_dst, ebuf, ssum, E);
    k_edge_scatter<<<cdiv((long)E * 64, TB), TB, 0, stream>>>(ei_src, ei_dst, ebuf, hs, acc, E);
    // residual + gelu + layernorm -> f16
    k_post_ln<<<cdiv((long)Ng * 32, 128), 128, 0, stream>>>(acc, bias_mg, h_gene, gamma, beta, LNh, Ng);
    // 4) gathered gene_emb = LN[gene_idx] @ Wo + bo  [f16]
    gemm(LNh, WoT, bo, nullptr, geneGh, gene_idx, B, od, hid, 0);
    // 5) t = gene_emb_gathered @ Wbil  [f32]
    gemm(geneGh, WbilT, nullptr, tbuf, nullptr, nullptr, B, od, od, 0);
    // 6) d1 = gelu(drug_x @ Wd1 + bd1)  [f16]
    gemm(drugA, Wd1T, bd1, nullptr, d1h, nullptr, Ndr, hid, dd, 1);
    // 7) drug_emb = d1 @ Wd2 + bd2  [f32]
    gemm(d1h, Wd2T, bd2, drugE, nullptr, nullptr, Ndr, od, hid, 0);
    // 8) score = (t * drug_emb[drug_idx]).sum(-1) + bbil
    k_final<<<cdiv(B, TB), TB, 0, stream>>>(tbuf, drugE, drug_idx, bbil, out, B);
}
